// LogicGatedSNN_33947421507663
// MI455X (gfx1250) — matrix operand order repaired and checked by reading, not compiled, not run on hardware
//
#include <hip/hip_runtime.h>

typedef __attribute__((ext_vector_type(16))) _Float16 v16h;
typedef __attribute__((ext_vector_type(8)))  float    v8f;
typedef __attribute__((ext_vector_type(4)))  float    v4f;

#define IN_F    8192
#define OUT_F   8192
#define NSPLIT  8
#define KC      (IN_F / NSPLIT)   // 1024 K per wave
#define NTILES  (OUT_F / 16)      // 512 row-tiles
#define THRESH  50.0f
#define TRACE_DECAY 0.8f

__device__ __forceinline__ _Float16 wbit(float v) {
  return v > THRESH ? (_Float16)1.0f : (_Float16)0.0f;
}

// ---------------- kernel 0: spike input f32 -> f16 (16 KB, cached) ----------
__global__ void snn_xcvt(const float* __restrict__ x, _Float16* __restrict__ xh) {
  int i = blockIdx.x * blockDim.x + threadIdx.x;
  if (i < IN_F) xh[i] = (_Float16)x[i];   // x is exactly 0.0 or 1.0
}

// ---------------- kernel 1: binary-weight GEMV via v_wmma_f32_16x16x32_f16 --
// wave handles 16 output rows x KC inputs; B = x-chunk broadcast to all 16
// columns so every column of D holds the 16 partial dot products.
__global__ __launch_bounds__(256) void snn_gemv(const float* __restrict__ W,
                                                const _Float16* __restrict__ xh,
                                                float* __restrict__ partials) {
  const int lane  = threadIdx.x & 31;
  const int wid   = blockIdx.x * (blockDim.x >> 5) + (threadIdx.x >> 5);
  const int tile  = wid >> 3;          // / NSPLIT
  const int split = wid & (NSPLIT - 1);
  const int r0    = tile << 4;
  const int kb    = split * KC;
  const int hi    = lane >> 4;         // 0 = low half-wave, 1 = high
  const int koffA = hi * 16;           // A: K 0-15 vs 16-31
  const int koffB = hi * 16;           // B: K 0-15 vs 16-31

  const float* Wrow = W + (size_t)(r0 + (lane & 15)) * IN_F;

  v8f c = {0.f, 0.f, 0.f, 0.f, 0.f, 0.f, 0.f, 0.f};

  for (int k = kb; k < kb + KC; k += 32) {
    // 16 contiguous weights per lane in four groups of 4 (coalesced b128s)
    const v4f f0 = *(const v4f*)(Wrow + k + koffA);
    const v4f f1 = *(const v4f*)(Wrow + k + koffA + 4);
    const v4f f2 = *(const v4f*)(Wrow + k + koffA + 8);
    const v4f f3 = *(const v4f*)(Wrow + k + koffA + 12);

    v16h a;
    a[0]  = wbit(f0[0]); a[1]  = wbit(f0[1]); a[2]  = wbit(f0[2]); a[3]  = wbit(f0[3]);
    a[4]  = wbit(f1[0]); a[5]  = wbit(f1[1]); a[6]  = wbit(f1[2]); a[7]  = wbit(f1[3]);
    a[8]  = wbit(f2[0]); a[9]  = wbit(f2[1]); a[10] = wbit(f2[2]); a[11] = wbit(f2[3]);
    a[12] = wbit(f3[0]); a[13] = wbit(f3[1]); a[14] = wbit(f3[2]); a[15] = wbit(f3[3]);

    // broadcast x-chunk: halves are K-contiguous, 32B-aligned
    v16h b = *(const v16h*)(xh + k + koffB);

    c = __builtin_amdgcn_wmma_f32_16x16x32_f16(
        /*neg_a=*/false, a, /*neg_b=*/false, b,
        /*c_mod=*/(short)0, c, /*reuse_a=*/false, /*reuse_b=*/false);
  }

  // D column 0: lane 0 holds rows 0..7 (VGPR g -> M=g), lane 16 rows 8..15
  if ((lane & 15) == 0) {
    float* dst = partials + ((size_t)tile * NSPLIT + split) * 16 + hi * 8;
#pragma unroll
    for (int g = 0; g < 8; ++g) dst[g] = c[g];
  }
}

// ---------------- kernel 2: reduce K-splits + LIF, emit spikes & new_v -----
__global__ void snn_lif(const float* __restrict__ partials,
                        const float* __restrict__ mp,
                        const float* __restrict__ thr,
                        float* __restrict__ out) {
  int o = blockIdx.x * blockDim.x + threadIdx.x;
  if (o >= OUT_F) return;
  int tile = o >> 4, row = o & 15;
  const float* p = partials + (size_t)tile * NSPLIT * 16 + row;
  float cur = 0.f;
#pragma unroll
  for (int s = 0; s < NSPLIT; ++s) cur += p[s * 16];
  float v  = mp[o] * 0.7f + cur;
  float sp = (v >= thr[o]) ? 1.0f : 0.0f;
  out[o]         = sp;                       // spikes
  out[OUT_F + o] = v * (1.0f - sp) * 0.5f;   // new_v_mem
}

// ---------------- kernel 3: trace = trace*0.8 + outer(spikes, x) -----------
// Pure streaming (512 MB): non-temporal on the big trace stream.
__global__ __launch_bounds__(256) void snn_trace(const v4f* __restrict__ trace4,
                                                 const v4f* __restrict__ x4,
                                                 const float* __restrict__ spikes,
                                                 v4f* __restrict__ out4) {
  const size_t N4 = (size_t)OUT_F * IN_F / 4;
  const size_t stride = (size_t)gridDim.x * blockDim.x;
  for (size_t idx = (size_t)blockIdx.x * blockDim.x + threadIdx.x; idx < N4; idx += stride) {
    size_t elem = idx * 4;
    int o  = (int)(elem >> 13);          // row (IN_F = 8192 = 2^13)
    int i4 = (int)((elem & (IN_F - 1)) >> 2);
    float s = spikes[o];                 // uniform across most of a wave
    v4f t  = __builtin_nontemporal_load(&trace4[idx]);
    v4f xv = x4[i4];
    v4f r;
    r[0] = t[0] * TRACE_DECAY + s * xv[0];
    r[1] = t[1] * TRACE_DECAY + s * xv[1];
    r[2] = t[2] * TRACE_DECAY + s * xv[2];
    r[3] = t[3] * TRACE_DECAY + s * xv[3];
    __builtin_nontemporal_store(r, &out4[idx]);
  }
}

extern "C" void kernel_launch(void* const* d_in, const int* in_sizes, int n_in,
                              void* d_out, int out_size, void* d_ws, size_t ws_size,
                              hipStream_t stream) {
  (void)in_sizes; (void)n_in; (void)out_size; (void)ws_size;
  const float* x     = (const float*)d_in[0];  // spike_input        [8192]
  const float* W     = (const float*)d_in[1];  // synapse_states     [8192*8192]
  const float* mp    = (const float*)d_in[2];  // membrane_potential [8192]
  const float* thr   = (const float*)d_in[3];  // adaptive_threshold [8192]
  const float* trace = (const float*)d_in[4];  // eligibility_trace  [8192*8192]
  float* out = (float*)d_out;   // [spikes 8192 | new_v 8192 | new_trace 8192*8192]

  // workspace: partials (512*8*16 f32 = 256 KB) then xh (8192 f16 = 16 KB)
  float*     partials = (float*)d_ws;
  _Float16*  xh = (_Float16*)((char*)d_ws + (size_t)NTILES * NSPLIT * 16 * sizeof(float));

  snn_xcvt<<<IN_F / 256, 256, 0, stream>>>(x, xh);
  snn_gemv<<<(NTILES * NSPLIT) / 8, 256, 0, stream>>>(W, xh, partials);
  snn_lif<<<OUT_F / 256, 256, 0, stream>>>(partials, mp, thr, out);
  snn_trace<<<16384, 256, 0, stream>>>((const v4f*)trace, (const v4f*)x,
                                       out /*spikes*/, (v4f*)(out + 2 * OUT_F));
}
